// GraphLayer_51848845197772
// MI455X (gfx1250) — compile-verified, matrix operand order checked by reading, not run on hardware
//
#include <hip/hip_runtime.h>

// ---------------------------------------------------------------------------
// Transformer block for MI455X (gfx1250, wave32, WMMA).
// GEMMs: v_wmma_f32_16x16x32_bf16, B-panels staged to LDS via
// GLOBAL_LOAD_ASYNC_TO_LDS_B128 (ASYNCcnt), A fragments register-pipelined
// (2-deep, no register rotation), B fragments group-loaded per k-step.
// Shapes: B=4, N=1024, D_MODEL=512, H=8, DH=64, D_INNER=2048.
// ---------------------------------------------------------------------------

#define D_MODEL 512
#define NSEQ    1024
#define NB      4
#define NH      8
#define DHEAD   64
#define D_INNER 2048
#define MROWS   (NB * NSEQ)      // 4096 token rows
#define KC      512              // K-chunk staged in LDS (64 x 512 bf16 = 64KB)

typedef __attribute__((ext_vector_type(16))) __bf16 v16bf;
typedef __attribute__((ext_vector_type(8)))  float  v8f;

// Pointer types for the async-to-LDS builtin: 4 x i32 vector pointee,
// global (AS1) source / LDS (AS3) destination.
typedef int v4i __attribute__((vector_size(16)));
typedef __attribute__((address_space(1))) v4i gv4i;
typedef __attribute__((address_space(3))) v4i sv4i;

__device__ __forceinline__ v8f zero8() {
  v8f z;
#pragma unroll
  for (int i = 0; i < 8; ++i) z[i] = 0.0f;
  return z;
}

// A/B fragment loader for 16-bit 16x32 WMMA operands.
// ISA layout: lanes 0-15 hold K=0..7 and 16..23; lanes 16-31 hold K=8..15 and 24..31.
__device__ __forceinline__ v16bf load_frag_bf(const __bf16* __restrict__ P, int ld,
                                              int r0, int k0, int lane) {
  const int row = r0 + (lane & 15);
  const int kb  = k0 + ((lane >> 4) << 3);
  const __bf16* p = P + (size_t)row * ld + kb;
  v16bf f;
#pragma unroll
  for (int j = 0; j < 8; ++j) { f[j] = p[j]; f[j + 8] = p[j + 16]; }
  return f;
}

// Same pattern from fp32 LDS (softmax probabilities), convert on load.
__device__ __forceinline__ v16bf load_frag_lds_f32(const float* __restrict__ S, int ld,
                                                   int k0, int lane) {
  const int row = lane & 15;
  const int kb  = k0 + ((lane >> 4) << 3);
  const float* p = S + (size_t)row * ld + kb;
  v16bf f;
#pragma unroll
  for (int j = 0; j < 8; ++j) { f[j] = (__bf16)p[j]; f[j + 8] = (__bf16)p[j + 16]; }
  return f;
}

__device__ __forceinline__ v8f wmma_bf16(v16bf a, v16bf b, v8f c) {
  return __builtin_amdgcn_wmma_f32_16x16x32_bf16(false, a, false, b, (short)0, c,
                                                 false, false);
}

__device__ __forceinline__ void wait_async_all() {
#if __has_builtin(__builtin_amdgcn_s_wait_asynccnt)
  __builtin_amdgcn_s_wait_asynccnt(0);
#endif
}

// Stage a 64 x KC bf16 B panel (rows n0..n0+63 of W[n,k] row-major) into LDS.
// 256 threads, 16 bytes per thread per iteration. Async direct-to-LDS when the
// gfx1250 builtin is available (no VGPR round-trip, tracked by ASYNCcnt).
__device__ __forceinline__ void stage_B_panel(const __bf16* __restrict__ W, int ldb,
                                              int n0, int kc0,
                                              __bf16* __restrict__ lds) {
  const int tid = threadIdx.x;
#pragma unroll
  for (int base = tid * 8; base < 64 * KC; base += 256 * 8) {
    const int r = base >> 9;          // base / KC
    const int c = base & (KC - 1);    // base % KC
    const __bf16* src = W + (size_t)(n0 + r) * ldb + kc0 + c;
#if __has_builtin(__builtin_amdgcn_global_load_async_to_lds_b128)
    __builtin_amdgcn_global_load_async_to_lds_b128(
        (gv4i*)src, (sv4i*)(lds + base), 0, 0);
#else
    *(int4*)(lds + base) = *(const int4*)src;
#endif
  }
}

// Issue 4 WMMAs against one A fragment; B fragments are group-loaded first so
// their ds_loads pipeline and a single dscnt wait covers the whole group.
__device__ __forceinline__ void wmma_group4(const __bf16* __restrict__ lds_b,
                                            int k0, int lane, v16bf a, v8f acc[4]) {
  v16bf b0 = load_frag_bf(lds_b, KC, 0,  k0, lane);
  v16bf b1 = load_frag_bf(lds_b, KC, 16, k0, lane);
  v16bf b2 = load_frag_bf(lds_b, KC, 32, k0, lane);
  v16bf b3 = load_frag_bf(lds_b, KC, 48, k0, lane);
  acc[0] = wmma_bf16(a, b0, acc[0]);
  acc[1] = wmma_bf16(a, b1, acc[1]);
  acc[2] = wmma_bf16(a, b2, acc[2]);
  acc[3] = wmma_bf16(a, b3, acc[3]);
}

// Block-tiled GEMM core: 8 waves, wave w owns rows m0 = bm*128 + w*16, all
// waves share the 64-wide N panel staged in LDS. k-loop unrolled by 2 with two
// live A fragments (a0 loop-carried, a1 local) -> no register rotation moves.
template <int KTOT>
__device__ __forceinline__ void block_gemm_16x64(const __bf16* __restrict__ A, int lda,
                                                 const __bf16* __restrict__ W, int ldb,
                                                 __bf16* __restrict__ lds_b,
                                                 int m0, int n0, int lane, v8f acc[4]) {
  v16bf a0 = load_frag_bf(A, lda, m0, 0, lane);
  for (int kc0 = 0; kc0 < KTOT; kc0 += KC) {
    if (kc0) __syncthreads();                 // previous chunk fully consumed
    stage_B_panel(W, ldb, n0, kc0, lds_b);
    wait_async_all();
    __syncthreads();

    for (int k0 = 0; k0 < KC; k0 += 64) {
      v16bf a1 = load_frag_bf(A, lda, m0, kc0 + k0 + 32, lane);
      wmma_group4(lds_b, k0, lane, a0, acc);
      if (k0 + 64 < KC)
        a0 = load_frag_bf(A, lda, m0, kc0 + k0 + 64, lane);
      else if (kc0 + KC < KTOT)
        a0 = load_frag_bf(A, lda, m0, kc0 + KC, lane);
      wmma_group4(lds_b, k0 + 32, lane, a1, acc);
    }
  }
}

// ---------------------------------------------------------------------------
// fp32 -> bf16 weight conversion
// ---------------------------------------------------------------------------
__global__ void f2bf_kernel(const float* __restrict__ x, __bf16* __restrict__ y, int n) {
  int i = blockIdx.x * blockDim.x + threadIdx.x;
  if (i < n) y[i] = (__bf16)x[i];
}

// ---------------------------------------------------------------------------
// LayerNorm: one 128-thread block per row of 512, outputs bf16
// ---------------------------------------------------------------------------
__global__ __launch_bounds__(128) void ln_kernel(const float* __restrict__ X,
                                                 const float* __restrict__ g,
                                                 const float* __restrict__ bta,
                                                 __bf16* __restrict__ Y) {
  const int row = blockIdx.x;
  const float* x = X + (size_t)row * D_MODEL;
  const int t = threadIdx.x;
  float s = 0.0f, s2 = 0.0f;
#pragma unroll
  for (int i = t; i < D_MODEL; i += 128) { float v = x[i]; s += v; s2 += v * v; }
#pragma unroll
  for (int off = 16; off; off >>= 1) {
    s  += __shfl_xor(s,  off, 32);
    s2 += __shfl_xor(s2, off, 32);
  }
  __shared__ float r1[4], r2[4];
  const int w = t >> 5;
  if ((t & 31) == 0) { r1[w] = s; r2[w] = s2; }
  __syncthreads();
  s  = r1[0] + r1[1] + r1[2] + r1[3];
  s2 = r2[0] + r2[1] + r2[2] + r2[3];
  const float mu  = s * (1.0f / D_MODEL);
  const float var = s2 * (1.0f / D_MODEL) - mu * mu;
  const float rs  = rsqrtf(var + 1e-5f);
  __bf16* y = Y + (size_t)row * D_MODEL;
#pragma unroll
  for (int i = t; i < D_MODEL; i += 128)
    y[i] = (__bf16)((x[i] - mu) * rs * g[i] + bta[i]);
}

// ---------------------------------------------------------------------------
// QKV GEMM: Zn[4096x512] @ Wqkv^T + bqkv, scattered to q/k [B,H,N,DH] and
// vT [B,H,DH,N] (bf16). Feature f = h*192 + j; j<64 -> v, <128 -> q, else k.
// ---------------------------------------------------------------------------
__global__ __launch_bounds__(256) void qkv_gemm(const __bf16* __restrict__ A,
                                                const __bf16* __restrict__ W,
                                                const float* __restrict__ bias,
                                                __bf16* __restrict__ q,
                                                __bf16* __restrict__ kk,
                                                __bf16* __restrict__ vT) {
  __shared__ __bf16 Bpanel[64 * KC];
  const int lane = threadIdx.x & 31;
  const int w    = threadIdx.x >> 5;
  const int bm = blockIdx.x & 31;               // MROWS/128 = 32
  const int tn = blockIdx.x >> 5;
  const int m0 = bm * 128 + w * 16, n0 = tn * 64;
  v8f acc[4];
#pragma unroll
  for (int t = 0; t < 4; ++t) acc[t] = zero8();
  block_gemm_16x64<D_MODEL>(A, D_MODEL, W, D_MODEL, Bpanel, m0, n0, lane, acc);

  const int half = lane >> 4, col = lane & 15;
#pragma unroll
  for (int t = 0; t < 4; ++t) {
    const int n = n0 + 16 * t + col;
    const float bn = bias[n];
    const int h = n / (3 * DHEAD), j = n % (3 * DHEAD);
#pragma unroll
    for (int i = 0; i < 8; ++i) {
      const int m = m0 + i + 8 * half;
      const int b = m >> 10, r = m & 1023;
      const float val = acc[t][i] + bn;
      if (j < DHEAD) {
        vT[(((size_t)(b * NH + h) * DHEAD) + j) * NSEQ + r] = (__bf16)val;
      } else if (j < 2 * DHEAD) {
        q[(((size_t)(b * NH + h) * NSEQ) + r) * DHEAD + (j - DHEAD)] = (__bf16)val;
      } else {
        kk[(((size_t)(b * NH + h) * NSEQ) + r) * DHEAD + (j - 2 * DHEAD)] = (__bf16)val;
      }
    }
  }
}

// ---------------------------------------------------------------------------
// Attention: one wave per (b, h, 16-row tile).
//   S = QK^T/8 + new_mask - gamma*D  (fp32 strip in LDS, 16x1024 = 64KB)
//   row softmax (shuffle reductions), * mask
//   attn = P @ V via WMMA (P read from LDS as bf16 fragments)
//   LeakyReLU fused into bf16 store -> attn_out[4096, 512]
// ---------------------------------------------------------------------------
__global__ __launch_bounds__(32) void attn_kernel(const __bf16* __restrict__ q,
                                                  const __bf16* __restrict__ k,
                                                  const __bf16* __restrict__ vT,
                                                  const float* __restrict__ Dm,
                                                  const float* __restrict__ nmask,
                                                  const float* __restrict__ mask,
                                                  const float* __restrict__ gamma_p,
                                                  __bf16* __restrict__ attn_out) {
  __shared__ float S[16][NSEQ];
  const int lane = threadIdx.x;
  const int blk = blockIdx.x;                   // b*8*64 + h*64 + tm
  const int tm = blk & 63;
  const int h  = (blk >> 6) & 7;
  const int b  = blk >> 9;
  const __bf16* qbh = q  + (size_t)(b * NH + h) * NSEQ * DHEAD;
  const __bf16* kbh = k  + (size_t)(b * NH + h) * NSEQ * DHEAD;
  const __bf16* vbh = vT + (size_t)(b * NH + h) * DHEAD * NSEQ;
  const int m0 = tm * 16;
  const float gamma = gamma_p[0];
  const int half = lane >> 4, col = lane & 15;

  // Q fragments are invariant across the column loop: hoist both k-steps.
  v16bf qf0 = load_frag_bf(qbh, DHEAD, m0, 0,  lane);
  v16bf qf1 = load_frag_bf(qbh, DHEAD, m0, 32, lane);

  for (int n0 = 0; n0 < NSEQ; n0 += 16) {
    v8f acc = zero8();
    v16bf kf0 = load_frag_bf(kbh, DHEAD, n0, 0,  lane);
    v16bf kf1 = load_frag_bf(kbh, DHEAD, n0, 32, lane);
    acc = wmma_bf16(qf0, kf0, acc);
    acc = wmma_bf16(qf1, kf1, acc);
    const int n = n0 + col;
#pragma unroll
    for (int i = 0; i < 8; ++i) {
      const int ml = i + 8 * half;
      const size_t idx = ((size_t)(b * NSEQ) + (m0 + ml)) * NSEQ + n;
      S[ml][n] = acc[i] * 0.125f + nmask[idx] - gamma * Dm[idx];
    }
  }
  __syncthreads();

  for (int r = 0; r < 16; ++r) {
    float mx = -1e30f;
    for (int c = lane; c < NSEQ; c += 32) mx = fmaxf(mx, S[r][c]);
#pragma unroll
    for (int off = 16; off; off >>= 1) mx = fmaxf(mx, __shfl_xor(mx, off, 32));
    float sum = 0.0f;
    for (int c = lane; c < NSEQ; c += 32) {
      float e = __expf(S[r][c] - mx);
      S[r][c] = e;
      sum += e;
    }
#pragma unroll
    for (int off = 16; off; off >>= 1) sum += __shfl_xor(sum, off, 32);
    const float inv = 1.0f / sum;
    const float* mrow = mask + ((size_t)(b * NSEQ) + (m0 + r)) * NSEQ;
    for (int c = lane; c < NSEQ; c += 32) S[r][c] = S[r][c] * inv * mrow[c];
  }
  __syncthreads();

  v8f acc[4];
#pragma unroll
  for (int t = 0; t < 4; ++t) acc[t] = zero8();
  for (int k0 = 0; k0 < NSEQ; k0 += 32) {
    v16bf af = load_frag_lds_f32(&S[0][0], NSEQ, k0, lane);
    // Group-load the 4 V fragments, then 4 WMMAs (row = d in vT layout).
    v16bf b0 = load_frag_bf(vbh, NSEQ, 0,  k0, lane);
    v16bf b1 = load_frag_bf(vbh, NSEQ, 16, k0, lane);
    v16bf b2 = load_frag_bf(vbh, NSEQ, 32, k0, lane);
    v16bf b3 = load_frag_bf(vbh, NSEQ, 48, k0, lane);
    acc[0] = wmma_bf16(af, b0, acc[0]);
    acc[1] = wmma_bf16(af, b1, acc[1]);
    acc[2] = wmma_bf16(af, b2, acc[2]);
    acc[3] = wmma_bf16(af, b3, acc[3]);
  }
#pragma unroll
  for (int t = 0; t < 4; ++t) {
    const int d = 16 * t + col;
#pragma unroll
    for (int i = 0; i < 8; ++i) {
      const int m = m0 + i + 8 * half;
      float v = acc[t][i];
      v = (v > 0.0f) ? v : 0.01f * v;              // leaky_relu fused
      attn_out[((size_t)(b * NSEQ) + m) * D_MODEL + h * DHEAD + d] = (__bf16)v;
    }
  }
}

// ---------------------------------------------------------------------------
// Wo GEMM: lrelu(attn)[4096x512] @ Wo^T + Z residual -> Z2 fp32
// ---------------------------------------------------------------------------
__global__ __launch_bounds__(256) void wo_gemm(const __bf16* __restrict__ A,
                                               const __bf16* __restrict__ W,
                                               const float* __restrict__ Zres,
                                               float* __restrict__ Z2) {
  __shared__ __bf16 Bpanel[64 * KC];
  const int lane = threadIdx.x & 31;
  const int w    = threadIdx.x >> 5;
  const int bm = blockIdx.x & 31;
  const int tn = blockIdx.x >> 5;
  const int m0 = bm * 128 + w * 16, n0 = tn * 64;
  v8f acc[4];
#pragma unroll
  for (int t = 0; t < 4; ++t) acc[t] = zero8();
  block_gemm_16x64<D_MODEL>(A, D_MODEL, W, D_MODEL, Bpanel, m0, n0, lane, acc);
  const int half = lane >> 4, col = lane & 15;
#pragma unroll
  for (int t = 0; t < 4; ++t) {
    const int n = n0 + 16 * t + col;
#pragma unroll
    for (int i = 0; i < 8; ++i) {
      const int m = m0 + i + 8 * half;
      const size_t idx = (size_t)m * D_MODEL + n;
      Z2[idx] = acc[t][i] + Zres[idx];
    }
  }
}

// ---------------------------------------------------------------------------
// FFN1 GEMM: Zn2[4096x512] @ Wp1^T + bp1, ReLU -> bf16
// ---------------------------------------------------------------------------
__global__ __launch_bounds__(256) void ffn1_gemm(const __bf16* __restrict__ A,
                                                 const __bf16* __restrict__ W,
                                                 const float* __restrict__ bias,
                                                 __bf16* __restrict__ H1) {
  __shared__ __bf16 Bpanel[64 * KC];
  const int lane = threadIdx.x & 31;
  const int w    = threadIdx.x >> 5;
  const int bm = blockIdx.x & 31;
  const int tn = blockIdx.x >> 5;
  const int m0 = bm * 128 + w * 16, n0 = tn * 64;
  v8f acc[4];
#pragma unroll
  for (int t = 0; t < 4; ++t) acc[t] = zero8();
  block_gemm_16x64<D_MODEL>(A, D_MODEL, W, D_MODEL, Bpanel, m0, n0, lane, acc);
  const int half = lane >> 4, col = lane & 15;
#pragma unroll
  for (int t = 0; t < 4; ++t) {
    const int n = n0 + 16 * t + col;
    const float bn = bias[n];
#pragma unroll
    for (int i = 0; i < 8; ++i) {
      const int m = m0 + i + 8 * half;
      float v = acc[t][i] + bn;
      v = (v > 0.0f) ? v : 0.0f;
      H1[(size_t)m * D_INNER + n] = (__bf16)v;
    }
  }
}

// ---------------------------------------------------------------------------
// FFN2 GEMM: H1[4096x2048] @ Wp2^T + bp2 + Z2 -> out fp32 (4 K-chunks)
// ---------------------------------------------------------------------------
__global__ __launch_bounds__(256) void ffn2_gemm(const __bf16* __restrict__ A,
                                                 const __bf16* __restrict__ W,
                                                 const float* __restrict__ bias,
                                                 const float* __restrict__ Z2,
                                                 float* __restrict__ out) {
  __shared__ __bf16 Bpanel[64 * KC];
  const int lane = threadIdx.x & 31;
  const int w    = threadIdx.x >> 5;
  const int bm = blockIdx.x & 31;
  const int tn = blockIdx.x >> 5;
  const int m0 = bm * 128 + w * 16, n0 = tn * 64;
  v8f acc[4];
#pragma unroll
  for (int t = 0; t < 4; ++t) acc[t] = zero8();
  block_gemm_16x64<D_INNER>(A, D_INNER, W, D_INNER, Bpanel, m0, n0, lane, acc);
  const int half = lane >> 4, col = lane & 15;
#pragma unroll
  for (int t = 0; t < 4; ++t) {
    const int n = n0 + 16 * t + col;
    const float bn = bias[n];
#pragma unroll
    for (int i = 0; i < 8; ++i) {
      const int m = m0 + i + 8 * half;
      const size_t idx = (size_t)m * D_MODEL + n;
      out[idx] = acc[t][i] + bn + Z2[idx];
    }
  }
}

// ---------------------------------------------------------------------------
// Launcher
// ---------------------------------------------------------------------------
extern "C" void kernel_launch(void* const* d_in, const int* in_sizes, int n_in,
                              void* d_out, int out_size, void* d_ws, size_t ws_size,
                              hipStream_t stream) {
  (void)in_sizes; (void)n_in; (void)out_size; (void)ws_size;
  const float* Z     = (const float*)d_in[0];
  const float* Dm    = (const float*)d_in[1];
  const float* nmask = (const float*)d_in[2];
  const float* mask  = (const float*)d_in[3];
  const float* Wqkv  = (const float*)d_in[4];
  const float* bqkv  = (const float*)d_in[5];
  const float* Wo    = (const float*)d_in[6];
  const float* g1    = (const float*)d_in[7];
  const float* b1    = (const float*)d_in[8];
  const float* g2    = (const float*)d_in[9];
  const float* b2    = (const float*)d_in[10];
  const float* Wp1   = (const float*)d_in[11];
  const float* bp1   = (const float*)d_in[12];
  const float* Wp2   = (const float*)d_in[13];
  const float* bp2   = (const float*)d_in[14];
  const float* gamma = (const float*)d_in[15];
  float* out = (float*)d_out;

  char* base = (char*)d_ws;
  size_t off = 0;
  auto bump = [&](size_t bytes) -> void* {
    void* p = base + off;
    off += (bytes + 255) & ~(size_t)255;
    return p;
  };
  __bf16* wqkv_bf = (__bf16*)bump((size_t)3 * D_MODEL * D_MODEL * 2);
  __bf16* wo_bf   = (__bf16*)bump((size_t)D_MODEL * D_MODEL * 2);
  __bf16* wp1_bf  = (__bf16*)bump((size_t)D_INNER * D_MODEL * 2);
  __bf16* wp2_bf  = (__bf16*)bump((size_t)D_MODEL * D_INNER * 2);
  __bf16* zn_bf   = (__bf16*)bump((size_t)MROWS * D_MODEL * 2);
  __bf16* q_bf    = (__bf16*)bump((size_t)NB * NH * NSEQ * DHEAD * 2);
  __bf16* k_bf    = (__bf16*)bump((size_t)NB * NH * NSEQ * DHEAD * 2);
  __bf16* vT_bf   = (__bf16*)bump((size_t)NB * NH * DHEAD * NSEQ * 2);
  __bf16* attn_bf = (__bf16*)bump((size_t)MROWS * D_MODEL * 2);
  float*  z2      = (float*) bump((size_t)MROWS * D_MODEL * 4);
  __bf16* zn2_bf  = (__bf16*)bump((size_t)MROWS * D_MODEL * 2);
  __bf16* h1_bf   = (__bf16*)bump((size_t)MROWS * D_INNER * 2);

  // Weight precision conversion (reused 4096x along M — pays for itself)
  {
    int n;
    n = 3 * D_MODEL * D_MODEL;
    f2bf_kernel<<<(n + 255) / 256, 256, 0, stream>>>(Wqkv, wqkv_bf, n);
    n = D_MODEL * D_MODEL;
    f2bf_kernel<<<(n + 255) / 256, 256, 0, stream>>>(Wo, wo_bf, n);
    n = D_INNER * D_MODEL;
    f2bf_kernel<<<(n + 255) / 256, 256, 0, stream>>>(Wp1, wp1_bf, n);
    n = D_MODEL * D_INNER;
    f2bf_kernel<<<(n + 255) / 256, 256, 0, stream>>>(Wp2, wp2_bf, n);
  }

  // LN1
  ln_kernel<<<MROWS, 128, 0, stream>>>(Z, g1, b1, zn_bf);

  // QKV projection: 32 m-blocks x 24 n64-panels
  qkv_gemm<<<32 * 24, 256, 0, stream>>>(zn_bf, wqkv_bf, bqkv, q_bf, k_bf, vT_bf);

  // Attention: one wave per (b, h, 16-row tile)
  attn_kernel<<<NB * NH * (NSEQ / 16), 32, 0, stream>>>(q_bf, k_bf, vT_bf, Dm, nmask,
                                                        mask, gamma, attn_bf);

  // Output projection + residual
  wo_gemm<<<32 * 8, 256, 0, stream>>>(attn_bf, wo_bf, Z, z2);

  // LN2
  ln_kernel<<<MROWS, 128, 0, stream>>>(z2, g2, b2, zn2_bf);

  // FFN
  ffn1_gemm<<<32 * 32, 256, 0, stream>>>(zn2_bf, wp1_bf, bp1, h1_bf);
  ffn2_gemm<<<32 * 8, 256, 0, stream>>>(h1_bf, wp2_bf, bp2, z2, out);
}
